// MUG_62105227100410
// MI455X (gfx1250) — matrix operand order repaired and checked by reading, not compiled, not run on hardware
//
#include <hip/hip_runtime.h>
#include <math.h>

// ---- WMMA vector types (gfx1250 CDNA5) ----
typedef __attribute__((ext_vector_type(16))) __bf16 v16bf;
typedef __attribute__((ext_vector_type(8)))  float  v8f;

#define HANC_M   3
#define HANC_H   8
#define HANC_D   64
#define HANC_F   512
#define HANC_SA  128
#define NEG_SLOPE 0.2f
#define BN_EPS    1e-5f

// ============================================================
// Generic utility kernels
// ============================================================
__global__ void fill_kernel(float* __restrict__ p, float v, long n) {
    long i = (long)blockIdx.x * blockDim.x + threadIdx.x;
    if (i < n) p[i] = v;
}

// pack two floats into one dword of 2 x bf16 (RNE) -> v_cvt_pk_bf16_f32
__device__ inline unsigned pk_bf16(float a, float b) {
    unsigned short lo = __builtin_bit_cast(unsigned short, (__bf16)a);
    unsigned short hi = __builtin_bit_cast(unsigned short, (__bf16)b);
    return (unsigned)lo | ((unsigned)hi << 16);
}

// ============================================================
// WMMA bf16 GEMM:  C[Mrows,Ncols] = A[Mrows,K] * B[K,Ncols]
// fp32 in/out, bf16 operands, fp32 accumulate.
// BM=128 (8 waves x 16 rows), BN=64 (4 wmma tiles per wave), BK=32.
// LDS tiles staged in WMMA *fragment order*: each lane's 16-bf16
// fragment is one contiguous 32B (2 x ds_load_b128) read.
// Staging is two-phase (all global loads first, then convert+store)
// with clamped addresses instead of exec-branching, so the 16 global
// loads per K-step pipeline instead of serializing on loadcnt.
// Requires: K % 32 == 0, Ncols % 64 == 0.  Mrows ragged (guarded).
// ============================================================
#define GBM 128
#define GBN 64
#define GBK 32
#define LROW 20   // row stride in dwords: 80B = 5*16B, conflict-free 20m mod 64

__global__ __launch_bounds__(256)
void gemm_bf16_wmma(const float* __restrict__ A, const float* __restrict__ B,
                    float* __restrict__ C, int Mrows, int Ncols, int Kdim) {
    alignas(16) __shared__ unsigned Apack[GBM * LROW];  // [m][half*8 + j/2]
    alignas(16) __shared__ unsigned Bpack[GBN * LROW];  // [n][half*8 + j/2]

    const int tid  = threadIdx.x;
    const int lane = tid & 31;
    const int wave = tid >> 5;        // 0..7 -> 16-row strip
    const int half = lane >> 4;       // 0/1
    const int l16  = lane & 15;
    const int blockM = blockIdx.x * GBM;
    const int blockN = blockIdx.y * GBN;

    // ---- loop-invariant staging offsets ----
    // A: element (r,c): lane-half=(c>>3)&1 ; frag slot j=(c&7)|((c>>1)&8)
    size_t aGlob[8];  int aLds[8];  float aMask[8];
    #pragma unroll
    for (int i = 0; i < 8; ++i) {
        int p  = i * 256 + tid;          // pair index 0..2047
        int r  = p >> 4;                 // row 0..127
        int c  = (p & 15) * 2;           // even col 0..30
        int gr = blockM + r;
        int cr = gr < Mrows ? gr : (Mrows - 1);     // clamp (no branch)
        aMask[i] = (gr < Mrows) ? 1.0f : 0.0f;
        aGlob[i] = (size_t)cr * Kdim + c;
        int ah = (c >> 3) & 1;
        int aj = (c & 7) | ((c >> 1) & 8);
        aLds[i] = r * LROW + ah * 8 + (aj >> 1);
    }
    // B: pair q (B[k0+2q][n], B[k0+2q+1][n]) stored in fragment order like A
    size_t bGlob[4];  int bLds[4];
    #pragma unroll
    for (int i = 0; i < 4; ++i) {
        int idx = i * 256 + tid;         // 0..1023
        int n   = idx & 63;
        int q   = idx >> 6;              // 0..15
        bGlob[i] = (size_t)(2 * q) * Ncols + blockN + n;
        bLds[i]  = n * LROW + ((q >> 2) & 1) * 8 + ((q & 3) | ((q >> 1) & 4));
    }
    const int aFrag = (wave * 16 + l16) * LROW + half * 8;
    int bFrag[4];
    #pragma unroll
    for (int t = 0; t < 4; ++t) bFrag[t] = (t * 16 + l16) * LROW + half * 8;

    v8f zero = {};
    v8f acc[4];
    for (int t = 0; t < 4; ++t) acc[t] = zero;

    for (int k0 = 0; k0 < Kdim; k0 += GBK) {
        // ---- phase 1: issue all global loads back-to-back ----
        float2 va[8];
        #pragma unroll
        for (int i = 0; i < 8; ++i)
            va[i] = *(const float2*)(A + aGlob[i] + k0);
        float bx[4], by[4];
        const float* Bk = B + (size_t)k0 * Ncols;
        #pragma unroll
        for (int i = 0; i < 4; ++i) { bx[i] = Bk[bGlob[i]]; by[i] = Bk[bGlob[i] + Ncols]; }
        // prefetch next A slab (gfx1250 global_prefetch_b8)
        if (k0 + GBK < Kdim)
            __builtin_prefetch(A + aGlob[tid & 7] + k0 + GBK, 0, 1);

        // ---- phase 2: convert + pack + LDS store ----
        #pragma unroll
        for (int i = 0; i < 8; ++i)
            Apack[aLds[i]] = pk_bf16(va[i].x * aMask[i], va[i].y * aMask[i]);
        #pragma unroll
        for (int i = 0; i < 4; ++i)
            Bpack[bLds[i]] = pk_bf16(bx[i], by[i]);
        __syncthreads();

        // ---- fragment loads: 32 contiguous bytes per lane ----
        v16bf afrag = *(const v16bf*)&Apack[aFrag];
        #pragma unroll
        for (int t = 0; t < 4; ++t) {
            v16bf bfrag = *(const v16bf*)&Bpack[bFrag[t]];
            acc[t] = __builtin_amdgcn_wmma_f32_16x16x32_bf16(
                false, afrag, false, bfrag, (short)0, acc[t], false, false);
        }
        __syncthreads();
    }

    // store: D element r lives at row (r + 8*half), col l16 within 16x16 tile
    #pragma unroll
    for (int t = 0; t < 4; ++t) {
        int col = blockN + t * 16 + l16;
        #pragma unroll
        for (int r = 0; r < 8; ++r) {
            int row = blockM + wave * 16 + r + 8 * half;
            if (row < Mrows) C[(size_t)row * Ncols + col] = acc[t][r];
        }
    }
}

// ============================================================
// el/er: per (n,h) dot over D
// ============================================================
__global__ void el_er_kernel(const float* __restrict__ ft,
                             const float* __restrict__ al, const float* __restrict__ ar,
                             float* __restrict__ el, float* __restrict__ er, int N) {
    int idx = blockIdx.x * blockDim.x + threadIdx.x;
    if (idx >= N * HANC_H) return;
    int n = idx >> 3, h = idx & 7;
    const float* fp  = ft + (size_t)n * HANC_F + h * HANC_D;
    const float* alp = al + h * HANC_D;
    const float* arp = ar + h * HANC_D;
    float sl = 0.f, sr = 0.f;
    #pragma unroll 4
    for (int d = 0; d < HANC_D; ++d) { float v = fp[d]; sl += v * alp[d]; sr += v * arp[d]; }
    el[idx] = sl; er[idx] = sr;
}

// ============================================================
// Edge softmax: max / exp+sum / aggregate
// ============================================================
__device__ inline void atomicMaxFloat(float* addr, float val) {
    unsigned int* a = (unsigned int*)addr;
    unsigned int cur = __float_as_uint(*addr);
    while (__uint_as_float(cur) < val) {
        unsigned int assumed = cur;
        cur = atomicCAS(a, assumed, __float_as_uint(val));
        if (cur == assumed) break;
    }
}

__global__ void edge_max_kernel(const float* __restrict__ el, const float* __restrict__ er,
                                const int* __restrict__ src, const int* __restrict__ dst,
                                float* __restrict__ ee, float* __restrict__ emax, int E) {
    int idx = blockIdx.x * blockDim.x + threadIdx.x;
    if (idx >= E * HANC_H) return;
    int e = idx >> 3, h = idx & 7;
    int s = src[e], d = dst[e];
    float x = el[s * HANC_H + h] + er[d * HANC_H + h];
    x = x > 0.f ? x : NEG_SLOPE * x;                 // leaky relu
    ee[idx] = x;
    atomicMaxFloat(&emax[d * HANC_H + h], x);
}

__global__ void edge_exp_kernel(const int* __restrict__ dst,
                                const float* __restrict__ emax,
                                float* __restrict__ ee, float* __restrict__ den, int E) {
    int idx = blockIdx.x * blockDim.x + threadIdx.x;
    if (idx >= E * HANC_H) return;
    int e = idx >> 3, h = idx & 7;
    int d = dst[e];
    float v = __expf(ee[idx] - emax[d * HANC_H + h]);
    ee[idx] = v;
    atomicAdd(&den[d * HANC_H + h], v);
}

// one edge handled by 128 threads, 4 contiguous floats each (coalesced float4)
__global__ void edge_agg_kernel(const float* __restrict__ ee, const float* __restrict__ den,
                                const float* __restrict__ ft,
                                const int* __restrict__ src, const int* __restrict__ dst,
                                float* __restrict__ rstm, int E) {
    long gid = (long)blockIdx.x * blockDim.x + threadIdx.x;
    if (gid >= (long)E * 128) return;
    int e = (int)(gid >> 7);
    int q = (int)(gid & 127);
    int f = q * 4;
    int h = f >> 6;
    int s = src[e], d = dst[e];
    float alpha = ee[e * HANC_H + h] / den[d * HANC_H + h];
    const float4 fv = *(const float4*)(ft + (size_t)s * HANC_F + f);
    float* o = rstm + (size_t)d * HANC_F + f;
    atomicAdd(o + 0, alpha * fv.x);
    atomicAdd(o + 1, alpha * fv.y);
    atomicAdd(o + 2, alpha * fv.z);
    atomicAdd(o + 3, alpha * fv.w);
}

// ============================================================
// residual + bias, BN stats, BN+ELU
// ============================================================
__global__ void resid_bias_kernel(float* __restrict__ rst, const float* __restrict__ hin,
                                  const float* __restrict__ b, int N) {
    long idx = (long)blockIdx.x * blockDim.x + threadIdx.x;
    long total = (long)HANC_M * N * HANC_F;
    if (idx >= total) return;
    int f = (int)(idx & (HANC_F - 1));
    int n = (int)((idx >> 9) % N);
    rst[idx] += hin[(size_t)n * HANC_F + f] + b[f];
}

__global__ __launch_bounds__(256)
void stats_kernel(const float* __restrict__ z, float* __restrict__ mu,
                  float* __restrict__ var, int N) {
    __shared__ float ssum[256], ssq[256];
    int b = blockIdx.x;                 // m*512 + f
    int m = b >> 9, f = b & (HANC_F - 1);
    const float* base = z + (size_t)m * N * HANC_F + f;
    float s = 0.f, q = 0.f;
    for (int n = threadIdx.x; n < N; n += 256) {
        float v = base[(size_t)n * HANC_F]; s += v; q += v * v;
    }
    ssum[threadIdx.x] = s; ssq[threadIdx.x] = q;
    __syncthreads();
    for (int st = 128; st > 0; st >>= 1) {
        if (threadIdx.x < st) { ssum[threadIdx.x] += ssum[threadIdx.x + st];
                                ssq[threadIdx.x]  += ssq[threadIdx.x + st]; }
        __syncthreads();
    }
    if (threadIdx.x == 0) {
        float mean = ssum[0] / N;
        mu[b] = mean;
        var[b] = ssq[0] / N - mean * mean;   // biased variance (matches jnp.var)
    }
}

__global__ void bn_elu_kernel(float* __restrict__ z, const float* __restrict__ mu,
                              const float* __restrict__ var, const float* __restrict__ g,
                              const float* __restrict__ be, int N) {
    long idx = (long)blockIdx.x * blockDim.x + threadIdx.x;
    long total = (long)HANC_M * N * HANC_F;
    if (idx >= total) return;
    int f = (int)(idx & (HANC_F - 1));
    int m = (int)(idx / ((long)N * HANC_F));
    int col = m * HANC_F + f;
    float v = z[idx];
    v = (v - mu[col]) * rsqrtf(var[col] + BN_EPS) * g[f] + be[f];
    z[idx] = v > 0.f ? v : expm1f(v);    // ELU
}

// ============================================================
// Semantic attention tail
// ============================================================
__global__ void wts_reduce_kernel(const float* __restrict__ tbuf,
                                  const float* __restrict__ sab,
                                  const float* __restrict__ sav,
                                  float* __restrict__ wsum, int N) {
    int r = blockIdx.x * blockDim.x + threadIdx.x;
    if (r >= HANC_M * N) return;
    const float* tp = tbuf + (size_t)r * HANC_SA;
    float s = 0.f;
    #pragma unroll 4
    for (int j = 0; j < HANC_SA; ++j) s += tanhf(tp[j] + sab[j]) * sav[j];
    atomicAdd(&wsum[r / N], s);
}

__global__ void beta_kernel(const float* __restrict__ wsum, float* __restrict__ beta,
                            float* att_out, int N) {
    float x[HANC_M];
    float mx = -1e30f;
    for (int m = 0; m < HANC_M; ++m) { x[m] = wsum[m] / (float)N; mx = fmaxf(mx, x[m]); }
    float den = 0.f;
    for (int m = 0; m < HANC_M; ++m) { x[m] = __expf(x[m] - mx); den += x[m]; }
    for (int m = 0; m < HANC_M; ++m) {
        float bm = x[m] / den;
        beta[m] = bm;
        if (att_out) att_out[m] = bm;
    }
}

__global__ void out_combine_kernel(const float* __restrict__ z, const float* __restrict__ beta,
                                   float* __restrict__ out, int N) {
    long idx = (long)blockIdx.x * blockDim.x + threadIdx.x;
    long NS = (long)N * HANC_F;
    if (idx >= NS) return;
    out[idx] = beta[0] * z[idx] + beta[1] * z[NS + idx] + beta[2] * z[2 * NS + idx];
}

// ============================================================
// Host-side orchestration for one HAN layer
// ============================================================
static inline int cdiv(long a, long b) { return (int)((a + b - 1) / b); }

static void run_layer(const float* hin, const int* es, const int* ed,
                      const float* W, const float* al, const float* ar,
                      const float* b, const float* g, const float* be,
                      const float* saW, const float* sab, const float* sav,
                      float* hout, float* att_out,
                      float* ft, float* el, float* er, float* emax, float* den,
                      float* ee, float* rst, float* tbuf, float* mu, float* var,
                      float* wsum, float* beta,
                      int N, int E, hipStream_t stream) {
    const int T = 256;
    // 1) projection GEMM: ft[N,512] = hin[N,512] @ W[512,512]   (WMMA bf16)
    {
        dim3 grid(cdiv(N, GBM), HANC_F / GBN);
        gemm_bf16_wmma<<<grid, T, 0, stream>>>(hin, W, ft, N, HANC_F, HANC_F);
    }
    // 2) el / er
    el_er_kernel<<<cdiv((long)N * HANC_H, T), T, 0, stream>>>(ft, al, ar, el, er, N);
    // 3) zero rst [M,N,512]
    fill_kernel<<<cdiv((long)HANC_M * N * HANC_F, T), T, 0, stream>>>(rst, 0.f, (long)HANC_M * N * HANC_F);
    // 4) per metapath graph
    for (int m = 0; m < HANC_M; ++m) {
        const int* sm = es + (size_t)m * E;
        const int* dm = ed + (size_t)m * E;
        float* rstm = rst + (size_t)m * N * HANC_F;
        fill_kernel<<<cdiv((long)N * HANC_H, T), T, 0, stream>>>(emax, -1e30f, (long)N * HANC_H);
        fill_kernel<<<cdiv((long)N * HANC_H, T), T, 0, stream>>>(den, 0.f, (long)N * HANC_H);
        edge_max_kernel<<<cdiv((long)E * HANC_H, T), T, 0, stream>>>(el, er, sm, dm, ee, emax, E);
        edge_exp_kernel<<<cdiv((long)E * HANC_H, T), T, 0, stream>>>(dm, emax, ee, den, E);
        edge_agg_kernel<<<cdiv((long)E * 128, T), T, 0, stream>>>(ee, den, ft, sm, dm, rstm, E);
    }
    // 5) residual + bias
    resid_bias_kernel<<<cdiv((long)HANC_M * N * HANC_F, T), T, 0, stream>>>(rst, hin, b, N);
    // 6) BN stats + normalize + ELU (in place -> z)
    stats_kernel<<<HANC_M * HANC_F, T, 0, stream>>>(rst, mu, var, N);
    bn_elu_kernel<<<cdiv((long)HANC_M * N * HANC_F, T), T, 0, stream>>>(rst, mu, var, g, be, N);
    // 7) semantic projection GEMM: tbuf[3N,128] = z[3N,512] @ saW[512,128]  (WMMA bf16)
    {
        dim3 grid(cdiv((long)HANC_M * N, GBM), HANC_SA / GBN);
        gemm_bf16_wmma<<<grid, T, 0, stream>>>(rst, saW, tbuf, HANC_M * N, HANC_SA, HANC_F);
    }
    // 8) tanh + dot(sav) + mean over nodes -> softmax beta
    fill_kernel<<<1, T, 0, stream>>>(wsum, 0.f, 16);
    wts_reduce_kernel<<<cdiv((long)HANC_M * N, T), T, 0, stream>>>(tbuf, sab, sav, wsum, N);
    beta_kernel<<<1, 1, 0, stream>>>(wsum, beta, att_out, N);
    // 9) beta-weighted combine -> hout[N,512]
    out_combine_kernel<<<cdiv((long)N * HANC_F, T), T, 0, stream>>>(rst, beta, hout, N);
}

// ============================================================
// Entry point
// ============================================================
extern "C" void kernel_launch(void* const* d_in, const int* in_sizes, int n_in,
                              void* d_out, int out_size, void* d_ws, size_t ws_size,
                              hipStream_t stream) {
    const float* h   = (const float*)d_in[0];
    const int*   es  = (const int*)d_in[1];
    const int*   ed  = (const int*)d_in[2];
    const float* W1  = (const float*)d_in[3];
    const float* al1 = (const float*)d_in[4];
    const float* ar1 = (const float*)d_in[5];
    const float* b1  = (const float*)d_in[6];
    const float* g1  = (const float*)d_in[7];
    const float* be1 = (const float*)d_in[8];
    const float* saW1= (const float*)d_in[9];
    const float* sab1= (const float*)d_in[10];
    const float* sav1= (const float*)d_in[11];
    const float* W2  = (const float*)d_in[12];
    const float* al2 = (const float*)d_in[13];
    const float* ar2 = (const float*)d_in[14];
    const float* b2  = (const float*)d_in[15];
    const float* g2  = (const float*)d_in[16];
    const float* be2 = (const float*)d_in[17];
    const float* saW2= (const float*)d_in[18];
    const float* sab2= (const float*)d_in[19];
    const float* sav2= (const float*)d_in[20];

    const int N = in_sizes[0] / HANC_F;
    const int E = in_sizes[1] / HANC_M;

    // workspace carve-up (floats)
    float* ws = (float*)d_ws;
    size_t off = 0;
    float* ft   = ws + off; off += (size_t)N * HANC_F;
    float* el   = ws + off; off += (size_t)N * HANC_H;
    float* er   = ws + off; off += (size_t)N * HANC_H;
    float* emax = ws + off; off += (size_t)N * HANC_H;
    float* den  = ws + off; off += (size_t)N * HANC_H;
    float* ee   = ws + off; off += (size_t)E * HANC_H;
    float* rst  = ws + off; off += (size_t)HANC_M * N * HANC_F;
    float* tbuf = ws + off; off += (size_t)HANC_M * N * HANC_SA;
    float* mu   = ws + off; off += (size_t)HANC_M * HANC_F;
    float* var  = ws + off; off += (size_t)HANC_M * HANC_F;
    float* wsum = ws + off; off += 16;
    float* beta = ws + off; off += 16;
    float* h1   = ws + off; off += (size_t)N * HANC_F;

    float* out  = (float*)d_out;              // h2 [N,512]
    float* att  = out + (size_t)N * HANC_F;   // att_mp [3]

    // layer 1: h -> h1 (beta discarded)
    run_layer(h, es, ed, W1, al1, ar1, b1, g1, be1, saW1, sab1, sav1,
              h1, nullptr,
              ft, el, er, emax, den, ee, rst, tbuf, mu, var, wsum, beta,
              N, E, stream);
    // layer 2: h1 -> out, beta -> att
    run_layer(h1, es, ed, W2, al2, ar2, b2, g2, be2, saW2, sab2, sav2,
              out, att,
              ft, el, er, emax, den, ee, rst, tbuf, mu, var, wsum, beta,
              N, E, stream);
}
